// LSHAttention_42434276885062
// MI455X (gfx1250) — matrix operand order repaired and checked by reading, not compiled, not run on hardware
//
#include <hip/hip_runtime.h>
#include <hip/hip_bf16.h>
#include <math.h>

typedef __attribute__((ext_vector_type(16))) _Float16 v16h;
typedef __attribute__((ext_vector_type(8)))  float    v8f;

#define B_  16
#define T_  4096
#define E_  256
#define NC_ 64      // n_chunks == BUCKET_SIZE
#define CL_ 64      // chunk_len = T/NC
#define H2_ 32      // BUCKET_SIZE/2

// ---------------------------------------------------------------------------
// Kernel 1: LSH bucketing.  buckets[b,t] = argmax([q.R, -q.R])
// ---------------------------------------------------------------------------
__global__ __launch_bounds__(256) void lsh_bucket_kernel(
    const float* __restrict__ q, const float* __restrict__ R,
    int* __restrict__ buckets) {
  __shared__ float Rs[E_ * H2_];
  const int b    = blockIdx.x / (T_ / 256);
  const int tblk = blockIdx.x % (T_ / 256);
  const float* Rb = R + (size_t)b * E_ * H2_;
  for (int i = threadIdx.x; i < E_ * H2_; i += 256) Rs[i] = Rb[i];
  __syncthreads();

  const int t = tblk * 256 + threadIdx.x;
  const float* qrow = q + ((size_t)b * T_ + t) * E_;
  float acc[H2_];
#pragma unroll
  for (int h = 0; h < H2_; ++h) acc[h] = 0.f;
  for (int e = 0; e < E_; ++e) {
    float qv = qrow[e];
    const float* rr = &Rs[e * H2_];
#pragma unroll
    for (int h = 0; h < H2_; ++h) acc[h] = fmaf(qv, rr[h], acc[h]);
  }
  float best = -INFINITY; int bi = 0;
#pragma unroll
  for (int idx = 0; idx < 2 * H2_; ++idx) {
    float v = (idx < H2_) ? acc[idx] : -acc[idx - H2_];
    if (v > best) { best = v; bi = idx; }   // strict > : first-index tie-break
  }
  buckets[(size_t)b * T_ + t] = bi;
}

// ---------------------------------------------------------------------------
// Kernel 2: per-batch histogram of 64 buckets + exclusive prefix sum
// ---------------------------------------------------------------------------
__global__ __launch_bounds__(256) void lsh_hist_kernel(
    const int* __restrict__ buckets, int* __restrict__ offsets) {
  __shared__ int cnt[NC_];
  const int b = blockIdx.x;
  if (threadIdx.x < NC_) cnt[threadIdx.x] = 0;
  __syncthreads();
  for (int t = threadIdx.x; t < T_; t += 256)
    atomicAdd(&cnt[buckets[(size_t)b * T_ + t]], 1);
  __syncthreads();
  if (threadIdx.x == 0) {
    int run = 0;
    for (int v = 0; v < NC_; ++v) { offsets[b * NC_ + v] = run; run += cnt[v]; }
  }
}

// ---------------------------------------------------------------------------
// Kernel 3: stable counting-sort scatter.  sticker[b,pos] = original token t.
// One thread per (batch, bucket); t ascending => stable (matches jnp.argsort).
// Lanes of a wave share b and read the same buckets[b,t] -> broadcast load.
// ---------------------------------------------------------------------------
__global__ __launch_bounds__(256) void lsh_scatter_kernel(
    const int* __restrict__ buckets, const int* __restrict__ offsets,
    int* __restrict__ sticker) {
  const int id = blockIdx.x * 256 + threadIdx.x;   // 0 .. B*NC-1
  const int b = id / NC_;
  const int v = id % NC_;
  int pos = offsets[b * NC_ + v];
  const int* brow = buckets + (size_t)b * T_;
  for (int t = 0; t < T_; ++t)
    if (brow[t] == v) sticker[(size_t)b * T_ + (pos++)] = t;
}

// ---------------------------------------------------------------------------
// Kernel 4: fused chunked attention.  One block = one (batch, chunk).
//   S = (cq . lbq^T)/sqrt(E) ; P = softmax(S) ; O = P . lbv ; scatter to out.
// ---------------------------------------------------------------------------
__global__ __launch_bounds__(256) void lsh_attn_kernel(
    const float* __restrict__ q, const float* __restrict__ val,
    const int* __restrict__ sticker, float* __restrict__ out) {
  __shared__ union {
    struct { _Float16 qa[CL_ * 32]; _Float16 qb[2 * CL_ * 32]; } p1;  // 12 KB
    _Float16 vt[CL_ * 2 * CL_];                                       // 16 KB
  } u;
  __shared__ float Ss[CL_ * 2 * CL_];     // 64 x 128 f32 scores      // 32 KB
  __shared__ int   tok[2 * CL_];

  const int b = blockIdx.x / NC_;
  const int c = blockIdx.x % NC_;
  const int tid  = threadIdx.x;
  const int lane = tid & 31;
  const int w    = tid >> 5;          // 8 waves
  const int hh   = lane >> 4;         // half-wave: 0/1
  const int r16  = lane & 15;

  // original token ids for the 128 look-one-back rows (first 64 == cq rows)
  if (tid < 128) {
    int j = tid;
    int pos = (j < CL_) ? (c * CL_ + j)
                        : (((c + NC_ - 1) % NC_) * CL_ + (j - CL_));
    tok[j] = sticker[(size_t)b * T_ + pos];
  }
  __syncthreads();

  // ---- Phase 1: S = cq . lbq^T, K streamed in 8 slices of 32 --------------
  const int mtile = w >> 1;           // 0..3  (S row tile)
  const int nbase = (w & 1) * 4;      // 0 or 4 (S col tile base, 4 tiles each)
  v8f acc[4];
#pragma unroll
  for (int n = 0; n < 4; ++n)
#pragma unroll
    for (int r = 0; r < 8; ++r) acc[n][r] = 0.f;

  const float* qbatch = q + (size_t)b * T_ * E_;
  for (int kk = 0; kk < 8; ++kk) {
    { // load cq slice 64x32 -> f16 LDS (8 contiguous floats / thread)
      int row = tid >> 2, colb = (tid & 3) * 8;
      const float* src = qbatch + (size_t)tok[row] * E_ + kk * 32 + colb;
      float4 f0 = ((const float4*)src)[0];
      float4 f1 = ((const float4*)src)[1];
      _Float16* dst = &u.p1.qa[row * 32 + colb];
      dst[0] = (_Float16)f0.x; dst[1] = (_Float16)f0.y;
      dst[2] = (_Float16)f0.z; dst[3] = (_Float16)f0.w;
      dst[4] = (_Float16)f1.x; dst[5] = (_Float16)f1.y;
      dst[6] = (_Float16)f1.z; dst[7] = (_Float16)f1.w;
    }
    { // load lbq slice 128x32 -> f16 LDS (16 contiguous floats / thread)
      int row = tid >> 1, colb = (tid & 1) * 16;
      const float* src = qbatch + (size_t)tok[row] * E_ + kk * 32 + colb;
      _Float16* dst = &u.p1.qb[row * 32 + colb];
#pragma unroll
      for (int i = 0; i < 4; ++i) {
        float4 f = ((const float4*)src)[i];
        dst[4 * i + 0] = (_Float16)f.x; dst[4 * i + 1] = (_Float16)f.y;
        dst[4 * i + 2] = (_Float16)f.z; dst[4 * i + 3] = (_Float16)f.w;
      }
    }
    __syncthreads();

    // A fragment: 16x32 f16 (ISA layout: lanes 0-15 K[0..7|16..23], 16-31 +8)
    v16h a;
    {
      const _Float16* p0 = &u.p1.qa[(mtile * 16 + r16) * 32 + hh * 8];
#pragma unroll
      for (int i = 0; i < 8; ++i) { a[i] = p0[i]; a[i + 8] = p0[16 + i]; }
    }
#pragma unroll
    for (int n = 0; n < 4; ++n) {
      // B fragment: B[k][n] = lbq[ncol][k]  (row of qb is contiguous in k)
      v16h bf;
      const _Float16* pb = &u.p1.qb[((nbase + n) * 16 + r16) * 32 + hh * 8];
#pragma unroll
      for (int i = 0; i < 8; ++i) { bf[i] = pb[i]; bf[i + 8] = pb[16 + i]; }
      acc[n] = __builtin_amdgcn_wmma_f32_16x16x32_f16(
          false, a, false, bf, (short)0, acc[n], false, false);
    }
    __syncthreads();
  }

  // write scaled scores (1/sqrt(256) = 0.0625)
#pragma unroll
  for (int n = 0; n < 4; ++n)
#pragma unroll
    for (int r = 0; r < 8; ++r) {
      int M = r + hh * 8;
      Ss[(mtile * 16 + M) * 128 + (nbase + n) * 16 + r16] = acc[n][r] * 0.0625f;
    }
  __syncthreads();

  // ---- Phase 2: row softmax in place (one thread per row) -----------------
  if (tid < CL_) {
    float* srow = &Ss[tid * 128];
    float mx = -INFINITY;
    for (int j = 0; j < 128; ++j) mx = fmaxf(mx, srow[j]);
    float sum = 0.f;
    for (int j = 0; j < 128; ++j) sum += __expf(srow[j] - mx);
    float inv = 1.f / sum;
    for (int j = 0; j < 128; ++j) srow[j] = __expf(srow[j] - mx) * inv;
  }
  __syncthreads();

  // ---- Phase 3: O = P . lbv, E streamed in 4 blocks of 64 cols ------------
  for (int eb = 0; eb < 4; ++eb) {
    { // load lbv block transposed: vt[e][j] (e:0..63 within block, j:0..127)
      int j = tid >> 1, ebase = (tid & 1) * 32;
      const float* src = val + ((size_t)b * T_ + tok[j]) * E_ + eb * 64 + ebase;
#pragma unroll
      for (int i = 0; i < 8; ++i) {
        float4 f = ((const float4*)src)[i];
        u.vt[(ebase + 4 * i + 0) * 128 + j] = (_Float16)f.x;
        u.vt[(ebase + 4 * i + 1) * 128 + j] = (_Float16)f.y;
        u.vt[(ebase + 4 * i + 2) * 128 + j] = (_Float16)f.z;
        u.vt[(ebase + 4 * i + 3) * 128 + j] = (_Float16)f.w;
      }
    }
    __syncthreads();

#pragma unroll
    for (int p = 0; p < 2; ++p) {       // 16 tiles (4 m x 4 n) / 8 waves
      int tt = w * 2 + p;
      int mt = tt >> 2, nt = tt & 3;
      v8f o;
#pragma unroll
      for (int r = 0; r < 8; ++r) o[r] = 0.f;
#pragma unroll
      for (int ks = 0; ks < 4; ++ks) {  // K = 128 in 4 slices of 32
        v16h a;                          // P rows, f32->f16 on the fly
        const float* p0 = &Ss[(mt * 16 + r16) * 128 + ks * 32 + hh * 8];
#pragma unroll
        for (int i = 0; i < 8; ++i) {
          a[i]     = (_Float16)p0[i];
          a[i + 8] = (_Float16)p0[16 + i];
        }
        v16h bf;                         // B[k][n] = lbv[k][ecol] = vt[ecol][k]
        const _Float16* pb = &u.vt[(nt * 16 + r16) * 128 + ks * 32 + hh * 8];
#pragma unroll
        for (int i = 0; i < 8; ++i) { bf[i] = pb[i]; bf[i + 8] = pb[16 + i]; }
        o = __builtin_amdgcn_wmma_f32_16x16x32_f16(
            false, a, false, bf, (short)0, o, false, false);
      }
      // scatter to un-sorted output position (undo_sort for free)
      int col = eb * 64 + nt * 16 + r16;
#pragma unroll
      for (int r = 0; r < 8; ++r) {
        int M = r + hh * 8;
        out[((size_t)b * T_ + tok[mt * 16 + M]) * E_ + col] = o[r];
      }
    }
    __syncthreads();
  }
}

// ---------------------------------------------------------------------------
extern "C" void kernel_launch(void* const* d_in, const int* in_sizes, int n_in,
                              void* d_out, int out_size, void* d_ws, size_t ws_size,
                              hipStream_t stream) {
  (void)in_sizes; (void)n_in; (void)out_size; (void)ws_size;
  const float* q = (const float*)d_in[0];
  // d_in[1] ("key") is unused by the reference computation.
  const float* v = (const float*)d_in[2];
  const float* R = (const float*)d_in[3];
  float* out = (float*)d_out;

  int* buckets = (int*)d_ws;                       // B*T ints
  int* offsets = buckets + (size_t)B_ * T_;        // B*NC ints
  int* sticker = offsets + B_ * NC_;               // B*T ints

  lsh_bucket_kernel<<<B_ * (T_ / 256), 256, 0, stream>>>(q, R, buckets);
  lsh_hist_kernel<<<B_, 256, 0, stream>>>(buckets, offsets);
  lsh_scatter_kernel<<<(B_ * NC_) / 256, 256, 0, stream>>>(buckets, offsets, sticker);
  lsh_attn_kernel<<<B_ * NC_, 256, 0, stream>>>(q, v, sticker, out);
}
